// MarginSoftmaxLoss_87729001988432
// MI455X (gfx1250) — compile-verified, matrix operand order checked by reading, not run on hardware
//
#include <hip/hip_runtime.h>
#include <hip/hip_bf16.h>
#include <string.h>

// ---------------------------------------------------------------------------
// MarginSoftmaxLoss (CosFace) for MI455X / gfx1250
//   B=512, D=256, C=200000, s=30, m=0.2
//
// Strategy:
//   * bf16 hi/lo split GEMM on v_wmma_f32_16x16x32_bf16 (3-term compensated:
//     ah*bh + ah*bl + al*bh  -> ~2^-16 relative accuracy on the cosine).
//   * logits are bounded by s=30, so logsumexp uses the fixed shift 30:
//     lse = 30 + log(sum exp(logit-30)).  One streaming pass, fused into the
//     GEMM epilogue; per-row partial sums combined with global atomic f32 add.
// ---------------------------------------------------------------------------

#define B_DIM 512
#define D_DIM 256
#define C_DIM 200000
#define S_SCALE 30.0f
#define MARGIN 0.2f

#define N_PER_BLOCK 2048                     // classes per block (8 waves * 16 tiles * 16)
#define N_BLOCKS ((C_DIM + N_PER_BLOCK - 1) / N_PER_BLOCK)   // 98
#define M_BLOCKS (B_DIM / 16)                                // 32

typedef __attribute__((ext_vector_type(16))) __bf16 v16bf;
typedef __attribute__((ext_vector_type(8)))  float  v8f;

// --------------------------- bf16 bit helpers ------------------------------
__device__ __forceinline__ unsigned short f2bf(float f) {
    unsigned u = __float_as_uint(f);
    u += 0x7FFFu + ((u >> 16) & 1u);     // round to nearest even
    return (unsigned short)(u >> 16);
}
__device__ __forceinline__ float bf2f(unsigned short s) {
    return __uint_as_float(((unsigned)s) << 16);
}

// ---- load one v16bf fragment: 8 bf16 at rowp[kbase..] + 8 at rowp[kbase+16..]
struct Bits256 { uint4 a, b; };
__device__ __forceinline__ v16bf load_frag(const unsigned short* rowp, int kbase) {
    Bits256 t;
    t.a = *(const uint4*)(rowp + kbase);
    t.b = *(const uint4*)(rowp + kbase + 16);
    return __builtin_bit_cast(v16bf, t);
}

// --------------------------- kernel 0: init --------------------------------
__global__ void init_bufs(float* rowSum, float* targetLogit) {
    int t = threadIdx.x;                 // 512 threads
    rowSum[t]      = 0.0f;
    targetLogit[t] = 0.0f;
}

// ---------------- kernel 1/2: row L2-normalize + bf16 hi/lo split ----------
// one block (256 threads) per row of length D=256
__global__ void __launch_bounds__(256)
normalize_split(const float* __restrict__ src,
                unsigned short* __restrict__ hi,
                unsigned short* __restrict__ lo) {
    __shared__ float red[256];
    const size_t row = blockIdx.x;
    const int tid = threadIdx.x;
    float v = src[row * D_DIM + tid];
    red[tid] = v * v;
    __syncthreads();
    for (int s = 128; s > 0; s >>= 1) {
        if (tid < s) red[tid] += red[tid + s];
        __syncthreads();
    }
    float inv = 1.0f / fmaxf(sqrtf(red[0]), 1e-12f);
    float xn = v * inv;
    unsigned short h = f2bf(xn);
    float r = xn - bf2f(h);
    hi[row * D_DIM + tid] = h;
    lo[row * D_DIM + tid] = f2bf(r);
}

// --------------- kernel 3: WMMA GEMM + fused exp-sum epilogue --------------
// block = 256 threads = 8 waves.  grid = (98 class-chunks, 32 row-tiles).
// wave w handles classes [chunk + w*256, chunk + w*256 + 256) as 16 tiles of 16,
// reusing its resident 16x256 A tile (hi+lo fragments) across all of them.
__global__ void __launch_bounds__(256)
gemm_softmax(const unsigned short* __restrict__ xh,
             const unsigned short* __restrict__ xl,
             const unsigned short* __restrict__ wh,
             const unsigned short* __restrict__ wl,
             const int* __restrict__ targets,
             float* __restrict__ rowSum,
             float* __restrict__ targetLogit) {
    const int lane = threadIdx.x & 31;
    const int wave = threadIdx.x >> 5;
    const int mbase = blockIdx.y * 16;
    const int chunkBase = blockIdx.x * N_PER_BLOCK;

    const int half = lane >> 4;          // 0: K-offset 0, 1: K-offset 8
    const int l15  = lane & 15;          // M (for A) / N (for B) within tile
    const int koff = half * 8;

    // A tile: rows mbase..mbase+15, all 256 K, resident in VGPRs (hi + lo).
    const unsigned short* pax = xh + (size_t)(mbase + l15) * D_DIM + koff;
    const unsigned short* pal = xl + (size_t)(mbase + l15) * D_DIM + koff;
    v16bf Ah[8], Al[8];
#pragma unroll
    for (int k = 0; k < 8; ++k) {
        Ah[k] = load_frag(pax, k * 32);
        Al[k] = load_frag(pal, k * 32);
    }

    // Per-lane streaming-softmax accumulators: element i of the C layout is
    // row M = i + 8*half (lanes 0-15 hold M=0..7, lanes 16-31 hold M=8..15).
    float accRow[8];
    int   tg[8];
#pragma unroll
    for (int i = 0; i < 8; ++i) {
        accRow[i] = 0.0f;
        tg[i] = targets[mbase + 8 * half + i];
    }

    for (int t = 0; t < 16; ++t) {
        const int ntile = chunkBase + wave * 256 + t * 16;
        if (ntile >= C_DIM) break;                  // wave-uniform exit
        const int n = ntile + l15;                  // this lane's class column
        const bool valid = (n < C_DIM);
        const int nc = valid ? n : 0;               // clamp for safe loads
        const unsigned short* pbh = wh + (size_t)nc * D_DIM + koff;
        const unsigned short* pbl = wl + (size_t)nc * D_DIM + koff;

        v8f acc = {};
#pragma unroll
        for (int k = 0; k < 8; ++k) {
            v16bf Bh = load_frag(pbh, k * 32);
            v16bf Bl = load_frag(pbl, k * 32);
            acc = __builtin_amdgcn_wmma_f32_16x16x32_bf16(
                      false, Ah[k], false, Bh, (short)0, acc, false, false);
            acc = __builtin_amdgcn_wmma_f32_16x16x32_bf16(
                      false, Ah[k], false, Bl, (short)0, acc, false, false);
            acc = __builtin_amdgcn_wmma_f32_16x16x32_bf16(
                      false, Al[k], false, Bh, (short)0, acc, false, false);
        }

        // epilogue: margin at target, exp with fixed shift S_SCALE
#pragma unroll
        for (int i = 0; i < 8; ++i) {
            float logit = S_SCALE * acc[i];
            if (valid && tg[i] == n) {
                logit -= S_SCALE * MARGIN;
                targetLogit[mbase + 8 * half + i] = logit;  // unique writer
            }
            float e = valid ? __expf(logit - S_SCALE) : 0.0f;
            accRow[i] += e;
        }
    }

    // reduce the 16 class-columns (lanes within each half) and atomically
    // fold this wave's contribution into the per-row sums.
#pragma unroll
    for (int i = 0; i < 8; ++i) {
        float v = accRow[i];
        v += __shfl_xor(v, 1);
        v += __shfl_xor(v, 2);
        v += __shfl_xor(v, 4);
        v += __shfl_xor(v, 8);
        if (l15 == 0) atomicAdd(&rowSum[mbase + 8 * half + i], v);
    }
}

// --------------------------- kernel 4: finalize ----------------------------
__global__ void finalize_loss(const float* __restrict__ rowSum,
                              const float* __restrict__ targetLogit,
                              float* __restrict__ out) {
    __shared__ float red[B_DIM];
    const int tid = threadIdx.x;                 // 512 threads
    // lse = 30 + log(sum exp(logit - 30)) ; nll = lse - targetLogit
    red[tid] = S_SCALE + logf(rowSum[tid]) - targetLogit[tid];
    __syncthreads();
    for (int s = B_DIM / 2; s > 0; s >>= 1) {
        if (tid < s) red[tid] += red[tid + s];
        __syncthreads();
    }
    if (tid == 0) out[0] = red[0] * (1.0f / (float)B_DIM);
}

// ---------------------------------------------------------------------------
extern "C" void kernel_launch(void* const* d_in, const int* in_sizes, int n_in,
                              void* d_out, int out_size, void* d_ws, size_t ws_size,
                              hipStream_t stream) {
    const float* x       = (const float*)d_in[0];   // (512, 256, 1) f32
    const float* w       = (const float*)d_in[1];   // (200000, 256, 1) f32
    const int*   targets = (const int*)d_in[2];     // (512,) int
    float*       out     = (float*)d_out;

    char* ws = (char*)d_ws;
    size_t off = 0;
    auto carve = [&](size_t bytes) {
        void* p = ws + off;
        off = (off + bytes + 255) & ~(size_t)255;
        return p;
    };
    unsigned short* xh = (unsigned short*)carve((size_t)B_DIM * D_DIM * 2);
    unsigned short* xl = (unsigned short*)carve((size_t)B_DIM * D_DIM * 2);
    unsigned short* wh = (unsigned short*)carve((size_t)C_DIM * D_DIM * 2);
    unsigned short* wl = (unsigned short*)carve((size_t)C_DIM * D_DIM * 2);
    float* rowSum      = (float*)carve((size_t)B_DIM * 4);
    float* targetLogit = (float*)carve((size_t)B_DIM * 4);

    init_bufs<<<1, B_DIM, 0, stream>>>(rowSum, targetLogit);
    normalize_split<<<B_DIM, D_DIM, 0, stream>>>(x, xh, xl);
    normalize_split<<<C_DIM, D_DIM, 0, stream>>>(w, wh, wl);
    gemm_softmax<<<dim3(N_BLOCKS, M_BLOCKS), 256, 0, stream>>>(
        xh, xl, wh, wl, targets, rowSum, targetLogit);
    finalize_loss<<<1, B_DIM, 0, stream>>>(rowSum, targetLogit, out);
}
